// SwinTransformerBlock_336_39006892982681
// MI455X (gfx1250) — matrix operand order repaired and checked by reading, not compiled, not run on hardware
//
#include <hip/hip_runtime.h>
#include <hip/hip_bf16.h>

// ---------------- types ----------------
typedef _Float16 h16;
typedef __attribute__((ext_vector_type(16))) _Float16 v16h;
typedef __attribute__((ext_vector_type(8)))  _Float16 v8h;
typedef __attribute__((ext_vector_type(8)))  float    v8f;

__device__ __forceinline__ v16h cat8(v8h lo, v8h hi) {
    return __builtin_shufflevector(lo, hi, 0,1,2,3,4,5,6,7,8,9,10,11,12,13,14,15);
}

// ---------------- problem constants ----------------
#define BB     32
#define HH     56
#define WW     56
#define CC     384
#define WS7    7
#define SHIFT3 3
#define HEADS  12
#define HD     32
#define NTOK   49         // tokens per window
#define HIDDEN 1536
#define NWIMG  64         // windows per image (8x8)
#define TOTWIN 2048       // BB * NWIMG
#define MTOK   100352     // BB*HH*WW  (rows of every big GEMM; 784*128)
#define SCALEQ 0.17677669529663687f   // 32^-0.5

// window-token row -> original (b,h,w) (also the reverse-roll mapping)
__device__ __forceinline__ void token_to_src(int r, int& b, int& h, int& w) {
    int win = r / NTOK, n = r % NTOK;
    b = win >> 6;
    int wrem = win & 63;
    int wh = wrem >> 3, ww = wrem & 7;
    int rr = n / 7, cc = n % 7;
    int hs = wh * 7 + rr, ws = ww * 7 + cc;
    h = (hs + SHIFT3) % HH;
    w = (ws + SHIFT3) % WW;
}

// ---------------- fp32 -> f16 conversion ----------------
__global__ void cvt_f16_kernel(const float* __restrict__ src, h16* __restrict__ dst, int n) {
    int i = blockIdx.x * blockDim.x + threadIdx.x;
    if (i < n) dst[i] = (h16)src[i];
}

// ---------------- LayerNorm (+optional shift/partition) ----------------
// mode 0: read x at rolled coords, output row = window-token order (for QKV A)
// mode 1: read src at natural token order (for FC1 A)
__global__ __launch_bounds__(384)
void ln_kernel(const float* __restrict__ src, const float* __restrict__ g,
               const float* __restrict__ bta, h16* __restrict__ dst, int mode) {
    int r = blockIdx.x;            // output row 0..MTOK-1
    int c = threadIdx.x;           // 0..383
    size_t sidx;
    if (mode == 0) {
        int b, h, w; token_to_src(r, b, h, w);
        sidx = ((((size_t)b * HH) + h) * WW + w) * CC + c;
    } else {
        sidx = (size_t)r * CC + c;
    }
    float v = src[sidx];
    float s = v, s2 = v * v;
    #pragma unroll
    for (int o = 16; o > 0; o >>= 1) { s += __shfl_xor(s, o); s2 += __shfl_xor(s2, o); }
    __shared__ float red[26];
    int wv = c >> 5, ln = c & 31;
    if (ln == 0) { red[wv] = s; red[13 + wv] = s2; }
    __syncthreads();
    if (c == 0) {
        float ts = 0.f, ts2 = 0.f;
        #pragma unroll
        for (int i = 0; i < 12; ++i) { ts += red[i]; ts2 += red[13 + i]; }
        red[12] = ts; red[25] = ts2;
    }
    __syncthreads();
    float mu  = red[12] * (1.0f / CC);
    float var = red[25] * (1.0f / CC) - mu * mu;
    float rs  = rsqrtf(var + 1e-5f);
    dst[(size_t)r * CC + c] = (h16)((v - mu) * rs * g[c] + bta[c]);
}

// ---------------- generic WMMA GEMM with fused epilogues ----------------
struct EpiArgs {
    const float* bias;
    float*       out;   // proj / fc2
    const float* xin;   // proj residual
    h16 *q, *k, *v;     // qkv scatter
    h16 *g;             // fc1 gelu out
};

// EPI: 0=QKV scatter, 1=PROJ(merge+roll+residual), 2=FC1+GELU, 3=FC2(+= out)
template <int EPI>
__global__ __launch_bounds__(256)
void gemm_kernel(const h16* __restrict__ A, const h16* __restrict__ Bm,
                 int K, int N, EpiArgs e) {
    __shared__ __align__(16) h16 As[128 * 40];   // [m][k], pitch 40 halves (80B)
    __shared__ __align__(16) h16 Bs[128 * 40];   // [n][k] transposed, pitch 40
    const int tid  = threadIdx.x;
    const int lane = tid & 31, wv = tid >> 5;
    const int lr = lane & 15, lh = lane >> 4;
    const int wm = (wv >> 1) * 32, wn = (wv & 1) * 64;
    const long blockM = (long)blockIdx.y * 128;
    const int  blockN = blockIdx.x * 128;

    v8f vzero = {};
    v8f acc[2][4];
    #pragma unroll
    for (int mi = 0; mi < 2; ++mi)
        #pragma unroll
        for (int ni = 0; ni < 4; ++ni) acc[mi][ni] = vzero;

    for (int k0 = 0; k0 < K; k0 += 32) {
        __syncthreads();
        // stage A tile 128x32 (two 16B chunks per thread)
        #pragma unroll
        for (int s = tid; s < 512; s += 256) {
            int row = s >> 2, q4 = s & 3;
            *(v8h*)&As[row * 40 + q4 * 8] =
                *(const v8h*)(A + (blockM + row) * (long)K + k0 + q4 * 8);
        }
        // stage B tile 32x128, transposed into [n][k]
        #pragma unroll
        for (int s = tid; s < 512; s += 256) {
            int kk = s >> 4, cg = s & 15;
            v8h bv = *(const v8h*)(Bm + (long)(k0 + kk) * N + blockN + cg * 8);
            #pragma unroll
            for (int t = 0; t < 8; ++t) Bs[(cg * 8 + t) * 40 + kk] = bv[t];
        }
        if (k0 + 32 < K) {  // CDNA5 global_prefetch for next tiles
            __builtin_prefetch(A + (blockM + (tid >> 1)) * (long)K + k0 + 32, 0, 1);
            __builtin_prefetch(Bm + (long)(k0 + 32 + (tid & 31)) * N + blockN, 0, 1);
        }
        __syncthreads();

        v16h af[2], bf[4];
        #pragma unroll
        for (int mi = 0; mi < 2; ++mi) {
            const h16* p = &As[(wm + mi * 16 + lr) * 40];
            af[mi] = cat8(*(const v8h*)(p + 8 * lh), *(const v8h*)(p + 16 + 8 * lh));
        }
        #pragma unroll
        for (int ni = 0; ni < 4; ++ni) {
            const h16* p = &Bs[(wn + ni * 16 + lr) * 40];
            bf[ni] = cat8(*(const v8h*)(p + 8 * lh), *(const v8h*)(p + 16 + 8 * lh));
        }
        #pragma unroll
        for (int mi = 0; mi < 2; ++mi)
            #pragma unroll
            for (int ni = 0; ni < 4; ++ni)
                acc[mi][ni] = __builtin_amdgcn_wmma_f32_16x16x32_f16(
                    false, af[mi], false, bf[ni], (short)0, acc[mi][ni], false, false);
    }

    // ---- fused epilogues ----
    #pragma unroll
    for (int mi = 0; mi < 2; ++mi)
        #pragma unroll
        for (int ni = 0; ni < 4; ++ni)
            #pragma unroll
            for (int i = 0; i < 8; ++i) {
                long gm = blockM + wm + mi * 16 + i + 8 * lh;
                int  gn = blockN + wn + ni * 16 + lr;
                float v = acc[mi][ni][i] + e.bias[gn];
                if constexpr (EPI == 0) {           // QKV scatter per (win,head)
                    int which = gn / CC, rem = gn % CC;
                    int head = rem >> 5, d = rem & 31;
                    long win = gm / NTOK, n = gm % NTOK;
                    h16* dst = (which == 0) ? e.q : (which == 1) ? e.k : e.v;
                    if (which == 0) v *= SCALEQ;
                    dst[((win * HEADS + head) * NTOK + n) * HD + d] = (h16)v;
                } else if constexpr (EPI == 1) {    // proj: merge + roll-back + residual
                    int b, h, w; token_to_src((int)gm, b, h, w);
                    long idx = (((long)b * HH + h) * WW + w) * CC + gn;
                    e.out[idx] = v + e.xin[idx];
                } else if constexpr (EPI == 2) {    // fc1 + exact GELU
                    float gl = 0.5f * v * (1.0f + erff(v * 0.70710678118654752f));
                    e.g[gm * (long)HIDDEN + gn] = (h16)gl;
                } else {                            // fc2 accumulate into x2
                    long idx = gm * (long)CC + gn;
                    e.out[idx] += v;
                }
            }
}

// ---------------- windowed attention: one wave per (window, head) ----------------
__global__ __launch_bounds__(128)
void attn_kernel(const h16* __restrict__ Qb, const h16* __restrict__ Kb,
                 const h16* __restrict__ Vb, const float* __restrict__ relt,
                 h16* __restrict__ Ob) {
    __shared__ __align__(16) h16  Pst[4 * 64 * 64];   // P (softmax out), per wave
    __shared__ __align__(16) h16  VTs[4 * 32 * 64];   // V^T zero-padded, per wave
    __shared__ float biasS[4 * 169];                  // rel-pos bias column, per wave
    const int win  = blockIdx.x;
    const int wv   = threadIdx.x >> 5, lane = threadIdx.x & 31;
    const int head = blockIdx.y * 4 + wv;
    const int lr = lane & 15, lh = lane >> 4;
    const int wh = (win & 63) >> 3, wwc = win & 7;
    const size_t base = ((size_t)win * HEADS + head) * NTOK * HD;

    // stage this head's relative-position bias (169 entries)
    for (int i = lane; i < 169; i += 32) biasS[wv * 169 + i] = relt[i * HEADS + head];
    // stage V^T [d][k_token], zero-padded to 64 tokens
    h16* vt = VTs + wv * 32 * 64;
    for (int i = lane; i < 32 * 64; i += 32) vt[i] = (h16)0.f;
    __syncthreads();
    for (int i = lane; i < NTOK * HD; i += 32) {
        int n = i >> 5, d = i & 31;
        vt[d * 64 + n] = Vb[base + i];
    }
    __syncthreads();

    // ---- S = (q*scale) @ k^T : 4x4 tiles of 16x16, K=32 ----
    v16h hzero = {};
    v16h aq[4];
    #pragma unroll
    for (int mt = 0; mt < 4; ++mt) {
        aq[mt] = hzero;
        int m = mt * 16 + lr;
        if (m < NTOK) {
            const h16* p = Qb + base + (size_t)m * HD;
            aq[mt] = cat8(*(const v8h*)(p + 8 * lh), *(const v8h*)(p + 16 + 8 * lh));
        }
    }
    v8f vzero = {};
    v8f sacc[4][4];
    #pragma unroll
    for (int mt = 0; mt < 4; ++mt)
        #pragma unroll
        for (int nt = 0; nt < 4; ++nt) sacc[mt][nt] = vzero;
    #pragma unroll
    for (int nt = 0; nt < 4; ++nt) {
        v16h bk = hzero;
        int n = nt * 16 + lr;
        if (n < NTOK) {
            const h16* p = Kb + base + (size_t)n * HD;
            bk = cat8(*(const v8h*)(p + 8 * lh), *(const v8h*)(p + 16 + 8 * lh));
        }
        #pragma unroll
        for (int mt = 0; mt < 4; ++mt)
            sacc[mt][nt] = __builtin_amdgcn_wmma_f32_16x16x32_f16(
                false, aq[mt], false, bk, (short)0, sacc[mt][nt], false, false);
    }

    // ---- bias + shift-mask + softmax in registers ----
    const float* bcol = biasS + wv * 169;
    #pragma unroll
    for (int mt = 0; mt < 4; ++mt)
        #pragma unroll
        for (int i = 0; i < 8; ++i) {
            int m = mt * 16 + i + 8 * lh;
            int ym = m / 7, xm = m % 7;
            int hm = wh * 7 + ym, wm_ = wwc * 7 + xm;
            int rm = (hm < 49 ? 0 : (hm < 53 ? 1 : 2)) * 3 +
                     (wm_ < 49 ? 0 : (wm_ < 53 ? 1 : 2));
            #pragma unroll
            for (int nt = 0; nt < 4; ++nt) {
                int n = nt * 16 + lr;
                float s = sacc[mt][nt][i];
                if (m < NTOK && n < NTOK) {
                    int yn = n / 7, xn = n % 7;
                    s += bcol[(ym - yn + 6) * 13 + (xm - xn + 6)];
                    int hn = wh * 7 + yn, wn_ = wwc * 7 + xn;
                    int rn = (hn < 49 ? 0 : (hn < 53 ? 1 : 2)) * 3 +
                             (wn_ < 49 ? 0 : (wn_ < 53 ? 1 : 2));
                    if (rm != rn) s -= 100.0f;
                } else s = -3.0e38f;
                sacc[mt][nt][i] = s;
            }
        }
    // softmax over each row (reduce over nt + 16 lanes of the half)
    #pragma unroll
    for (int mt = 0; mt < 4; ++mt)
        #pragma unroll
        for (int i = 0; i < 8; ++i) {
            float mx = -3.0e38f;
            #pragma unroll
            for (int nt = 0; nt < 4; ++nt) mx = fmaxf(mx, sacc[mt][nt][i]);
            #pragma unroll
            for (int o = 1; o < 16; o <<= 1) mx = fmaxf(mx, __shfl_xor(mx, o));
            float sum = 0.f;
            #pragma unroll
            for (int nt = 0; nt < 4; ++nt) {
                float ev = __expf(sacc[mt][nt][i] - mx);
                sacc[mt][nt][i] = ev; sum += ev;
            }
            #pragma unroll
            for (int o = 1; o < 16; o <<= 1) sum += __shfl_xor(sum, o);
            float inv = 1.0f / sum;
            #pragma unroll
            for (int nt = 0; nt < 4; ++nt) sacc[mt][nt][i] *= inv;
        }

    // ---- P -> LDS (f16), then O = P @ V ----
    h16* pst = Pst + wv * 64 * 64;
    #pragma unroll
    for (int mt = 0; mt < 4; ++mt)
        #pragma unroll
        for (int nt = 0; nt < 4; ++nt)
            #pragma unroll
            for (int i = 0; i < 8; ++i)
                pst[(mt * 16 + i + 8 * lh) * 64 + nt * 16 + lr] = (h16)sacc[mt][nt][i];
    __syncthreads();

    v8f oacc[4][2];
    #pragma unroll
    for (int mt = 0; mt < 4; ++mt) { oacc[mt][0] = vzero; oacc[mt][1] = vzero; }
    #pragma unroll
    for (int kt = 0; kt < 2; ++kt) {
        v16h ap[4];
        #pragma unroll
        for (int mt = 0; mt < 4; ++mt) {
            const h16* p = pst + (mt * 16 + lr) * 64 + kt * 32;
            ap[mt] = cat8(*(const v8h*)(p + 8 * lh), *(const v8h*)(p + 16 + 8 * lh));
        }
        #pragma unroll
        for (int nt = 0; nt < 2; ++nt) {
            const h16* p = vt + (nt * 16 + lr) * 64 + kt * 32;
            v16h bv = cat8(*(const v8h*)(p + 8 * lh), *(const v8h*)(p + 16 + 8 * lh));
            #pragma unroll
            for (int mt = 0; mt < 4; ++mt)
                oacc[mt][nt] = __builtin_amdgcn_wmma_f32_16x16x32_f16(
                    false, ap[mt], false, bv, (short)0, oacc[mt][nt], false, false);
        }
    }

    // ---- write O as f16 rows for the proj GEMM ----
    #pragma unroll
    for (int mt = 0; mt < 4; ++mt)
        #pragma unroll
        for (int nt = 0; nt < 2; ++nt)
            #pragma unroll
            for (int i = 0; i < 8; ++i) {
                int m = mt * 16 + i + 8 * lh;
                if (m < NTOK) {
                    int d = nt * 16 + lr;
                    Ob[((size_t)win * NTOK + m) * CC + head * HD + d] =
                        (h16)oacc[mt][nt][i];
                }
            }
}

// ---------------- host orchestration ----------------
extern "C" void kernel_launch(void* const* d_in, const int* in_sizes, int n_in,
                              void* d_out, int out_size, void* d_ws, size_t ws_size,
                              hipStream_t stream) {
    (void)in_sizes; (void)n_in; (void)out_size; (void)ws_size;
    const float* x     = (const float*)d_in[0];
    const float* n1g   = (const float*)d_in[1];
    const float* n1b   = (const float*)d_in[2];
    const float* qkvw  = (const float*)d_in[3];
    const float* qkvb  = (const float*)d_in[4];
    const float* projw = (const float*)d_in[5];
    const float* projb = (const float*)d_in[6];
    const float* relt  = (const float*)d_in[7];
    const float* n2g   = (const float*)d_in[8];
    const float* n2b   = (const float*)d_in[9];
    const float* fc1w  = (const float*)d_in[10];
    const float* fc1b  = (const float*)d_in[11];
    const float* fc2w  = (const float*)d_in[12];
    const float* fc2b  = (const float*)d_in[13];
    float* out = (float*)d_out;
    char*  ws  = (char*)d_ws;

    const size_t MB = 1ull << 20;
    // weights (f16) packed at front
    h16* wq16 = (h16*)(ws);
    h16* wp16 = (h16*)(ws + 884736);
    h16* w116 = (h16*)(ws + 884736 + 294912);
    h16* w216 = (h16*)(ws + 884736 + 294912 + 1179648);
    // activation regions (each 80MB; G reuses Q..O span 320MB)
    const size_t offA = 4 * MB;
    const size_t offQ = offA + 80 * MB;
    const size_t offK = offQ + 80 * MB;
    const size_t offV = offK + 80 * MB;
    const size_t offO = offV + 80 * MB;
    h16* Aw = (h16*)(ws + offA);   // LN1 wins -> later LN2 out
    h16* Qb = (h16*)(ws + offQ);
    h16* Kb = (h16*)(ws + offK);
    h16* Vb = (h16*)(ws + offV);
    h16* Ob = (h16*)(ws + offO);
    h16* Gb = (h16*)(ws + offQ);   // fc1/gelu out (308MB) reuses Q..O

    // 1) weight conversion
    cvt_f16_kernel<<<(442368 + 255) / 256, 256, 0, stream>>>(qkvw, wq16, 442368);
    cvt_f16_kernel<<<(147456 + 255) / 256, 256, 0, stream>>>(projw, wp16, 147456);
    cvt_f16_kernel<<<(589824 + 255) / 256, 256, 0, stream>>>(fc1w, w116, 589824);
    cvt_f16_kernel<<<(589824 + 255) / 256, 256, 0, stream>>>(fc2w, w216, 589824);

    // 2) LN1 + roll(-3,-3) + window partition -> f16
    ln_kernel<<<MTOK, 384, 0, stream>>>(x, n1g, n1b, Aw, 0);

    // 3) QKV GEMM [100352x384]@[384x1152] with q/k/v scatter
    EpiArgs eq = {}; eq.bias = qkvb; eq.q = Qb; eq.k = Kb; eq.v = Vb;
    gemm_kernel<0><<<dim3(1152 / 128, MTOK / 128), 256, 0, stream>>>(Aw, wq16, 384, 1152, eq);

    // 4) windowed attention (2048 windows x 12 heads, one wave/head)
    attn_kernel<<<dim3(TOTWIN, 3), 128, 0, stream>>>(Qb, Kb, Vb, relt, Ob);

    // 5) proj GEMM + merge + roll(+3,+3) + residual -> d_out (fp32 x2)
    EpiArgs ep = {}; ep.bias = projb; ep.out = out; ep.xin = x;
    gemm_kernel<1><<<dim3(384 / 128, MTOK / 128), 256, 0, stream>>>(Ob, wp16, 384, 384, ep);

    // 6) LN2 -> f16 (natural token order, reuse A region)
    ln_kernel<<<MTOK, 384, 0, stream>>>(out, n2g, n2b, Aw, 1);

    // 7) FC1 + GELU -> f16
    EpiArgs e1 = {}; e1.bias = fc1b; e1.g = Gb;
    gemm_kernel<2><<<dim3(1536 / 128, MTOK / 128), 256, 0, stream>>>(Aw, w116, 384, 1536, e1);

    // 8) FC2 accumulate into d_out
    EpiArgs e2 = {}; e2.bias = fc2b; e2.out = out;
    gemm_kernel<3><<<dim3(384 / 128, MTOK / 128), 256, 0, stream>>>(Gb, w216, 1536, 384, e2);
}
